// MultiheadAttention_81157702025959
// MI455X (gfx1250) — compile-verified, matrix operand order checked by reading, not run on hardware
//
#include <hip/hip_runtime.h>
#include <hip/hip_bf16.h>
#include <math.h>

// MultiheadAttention for MI455X (gfx1250, wave32, WMMA + TDM).
//   1) mha_proj_kernel (grid.z = 0/1/2 -> Q/K/V): C = A[4096x1024]*W[1024x1024]
//      fp32 in, bf16 out. Q,K row-major bf16; V stored transposed [b][h][d][s].
//   2) mha_attn_kernel: one wave per 16-query tile per (b,h); key blocks of 32.
//      K tiles streamed into LDS by the Tensor Data Mover (double buffered,
//      s_wait_tensorcnt pipelined); QK^T and PV on v_wmma_f32_16x16x32_bf16.

#define BATCH 2
#define SEQ   2048
#define DIM   1024
#define NH    16
#define HDIM  64

typedef __attribute__((ext_vector_type(16))) __bf16 v16bf;
typedef __attribute__((ext_vector_type(8)))  __bf16 v8bf;
typedef __attribute__((ext_vector_type(8)))  float  v8f;
typedef __attribute__((ext_vector_type(4)))  unsigned int v4u;
typedef __attribute__((ext_vector_type(8)))  int v8i;
typedef __attribute__((ext_vector_type(4)))  int v4i;

// Native bf16 conversion (let clang emit the hardware cvt op).
__device__ __forceinline__ __bf16 f2bf(float f) { return (__bf16)f; }

// Load a 16x32 bf16 fragment in the CDNA5 WMMA A/B lane layout:
//   lane L: row = L%16, holds K = kb..kb+7 and kb+16..kb+23, kb=(L/16)*8.
__device__ __forceinline__ v16bf load_frag(const __bf16* base, int stride) {
  const int L   = threadIdx.x & 31;
  const int row = L & 15;
  const int kb  = (L >> 4) << 3;
  const __bf16* p = base + (size_t)row * stride + kb;
  v8bf lo = *(const v8bf*)(p);
  v8bf hi = *(const v8bf*)(p + 16);
  v16bf r;
#pragma unroll
  for (int i = 0; i < 8; ++i) { r[i] = lo[i]; r[i + 8] = hi[i]; }
  return r;
}

// ---------------------------------------------------------------------------
// Tensor Data Mover: async-copy a 32(keys) x 64(dims) bf16 tile (row stride
// DIM elements) from global into LDS (packed rows of 64 elements).
// D# group0: count=1 | lds_addr | 57b global_addr | type=2.
// D# group1: data_size=1(2B), tensor 64x32, tile 64x32, dim0_stride=1024.
// ---------------------------------------------------------------------------
__device__ __forceinline__ void tdm_load_k_tile(const __bf16* gsrc, __bf16* ldst) {
  const unsigned long long ga = (unsigned long long)(size_t)gsrc;
  v4u g0;
  g0[0] = 1u;                                              // count=1, user D#
  g0[1] = (unsigned)__builtin_amdgcn_readfirstlane((int)(unsigned)(size_t)ldst);
  g0[2] = (unsigned)__builtin_amdgcn_readfirstlane((int)(unsigned)(ga & 0xFFFFFFFFull));
  g0[3] = ((unsigned)__builtin_amdgcn_readfirstlane(
               (int)(((unsigned)(ga >> 32)) & 0x01FFFFFFu))) |
          (2u << 30);                                      // type=2 ("image")
  v8i g1;
  g1[0] = 0x00010000;        // workgroup_mask=0, data_size=1 (2-byte elems)
  g1[1] = (HDIM & 0xFFFF) << 16;   // tensor_dim0 = 64 (low 16 bits)
  g1[2] = (32 << 16);              // tensor_dim0 hi=0 | tensor_dim1 = 32
  g1[3] = (HDIM << 16);            // tensor_dim1 hi=0 | tile_dim0 = 64
  g1[4] = 32;                      // tile_dim1 = 32, tile_dim2 = 0
  g1[5] = DIM;                     // tensor_dim0_stride = 1024 (low 32)
  g1[6] = 0;                       // stride0 hi | stride1 lo
  g1[7] = 0;                       // stride1 hi
  v4i z4 = {0, 0, 0, 0};
#if defined(__clang_major__) && (__clang_major__ >= 23)
  v8i z8 = {0, 0, 0, 0, 0, 0, 0, 0};
  __builtin_amdgcn_tensor_load_to_lds(g0, g1, z4, z4, z8, 0);
#else
  __builtin_amdgcn_tensor_load_to_lds(g0, g1, z4, z4, 0);
#endif
}

// ---------------------------------------------------------------------------
// Projection GEMM: block tile 128(M) x 64(N), 256 threads = 8 waves,
// wave w computes rows [w*16, w*16+16) x 4 N-tiles. K stepped by 32.
// ---------------------------------------------------------------------------
__global__ __launch_bounds__(256)
void mha_proj_kernel(const float* __restrict__ Q, const float* __restrict__ K,
                     const float* __restrict__ V, const float* __restrict__ Wq,
                     const float* __restrict__ Wk, const float* __restrict__ Wv,
                     __bf16* __restrict__ qws, __bf16* __restrict__ kws,
                     __bf16* __restrict__ vtws) {
  __shared__ __bf16 As[128 * 32];   // A tile, row-major [m][k]
  __shared__ __bf16 Bs[64 * 32];    // W tile transposed [n][k]

  const int z = blockIdx.z;
  const float* A = (z == 0) ? Q : (z == 1) ? K : V;
  const float* W = (z == 0) ? Wq : (z == 1) ? Wk : Wv;

  const int nBase = blockIdx.x * 64;
  const int mBase = blockIdx.y * 128;
  const int tid   = threadIdx.x;
  const int wave  = tid >> 5;
  const int lane  = tid & 31;

  v8f acc[4];
#pragma unroll
  for (int nt = 0; nt < 4; ++nt)
#pragma unroll
    for (int i = 0; i < 8; ++i) acc[nt][i] = 0.0f;

  for (int kBase = 0; kBase < DIM; kBase += 32) {
    {   // Stage A tile (128x32): thread t -> row t/2, half-row of 16 floats.
      const int row = tid >> 1;
      const int ks  = (tid & 1) * 16;
      const float* src = A + (size_t)(mBase + row) * DIM + kBase + ks;
#pragma unroll
      for (int i = 0; i < 16; ++i) As[row * 32 + ks + i] = f2bf(src[i]);
    }
    {   // Stage W tile transposed (store [n][k]): thread t -> k=t/8, 8 n's.
      const int kl = tid >> 3;
      const int nl = (tid & 7) * 8;
      const float* src = W + (size_t)(kBase + kl) * DIM + nBase + nl;
#pragma unroll
      for (int i = 0; i < 8; ++i) Bs[(nl + i) * 32 + kl] = f2bf(src[i]);
    }
    __syncthreads();

    v16bf fa = load_frag(&As[wave * 16 * 32], 32);
#pragma unroll
    for (int nt = 0; nt < 4; ++nt) {
      v16bf fb = load_frag(&Bs[nt * 16 * 32], 32);
      acc[nt] = __builtin_amdgcn_wmma_f32_16x16x32_bf16(
          false, fa, false, fb, (short)0, acc[nt], false, false);
    }
    __syncthreads();
  }

  // Epilogue. C layout: lane holds col = L%16; VGPR v holds row v (+8 hi half).
  const int colL    = lane & 15;
  const int rowHalf = (lane >> 4) << 3;
  if (z < 2) {
    __bf16* out = (z == 0) ? qws : kws;
#pragma unroll
    for (int nt = 0; nt < 4; ++nt) {
      const int col  = nBase + nt * 16 + colL;
      const int rowb = mBase + wave * 16 + rowHalf;
#pragma unroll
      for (int v = 0; v < 8; ++v)
        out[(size_t)(rowb + v) * DIM + col] = f2bf(acc[nt][v]);
    }
  } else {
    // V: store transposed -> vt[b][col=h*64+d][s], contiguous in s (=rows).
#pragma unroll
    for (int nt = 0; nt < 4; ++nt) {
      const int col = nBase + nt * 16 + colL;
      const int m0  = mBase + wave * 16 + rowHalf;   // flattened b*SEQ + s
      const int b   = m0 / SEQ;
      const int s0  = m0 % SEQ;
      v8bf pk;
#pragma unroll
      for (int v = 0; v < 8; ++v) pk[v] = f2bf(acc[nt][v]);
      *(v8bf*)(vtws + (size_t)b * DIM * SEQ + (size_t)col * SEQ + s0) = pk;
    }
  }
}

// ---------------------------------------------------------------------------
// Flash attention: 128 threads = 4 waves; each wave owns 16 query rows of one
// (b,h). Keys in blocks of 32, K tiles TDM-streamed into double-buffered LDS.
// ---------------------------------------------------------------------------
__global__ __launch_bounds__(128)
void mha_attn_kernel(const __bf16* __restrict__ qws,
                     const __bf16* __restrict__ kws,
                     const __bf16* __restrict__ vtws,
                     float* __restrict__ out) {
  __shared__ __bf16 ldsK[4][2][32 * 64];   // per-wave double-buffered K tile
  __shared__ __bf16 ldsP[4][16 * 32];      // per-wave P transpose staging

  const int tid  = threadIdx.x;
  const int wave = tid >> 5;
  const int lane = tid & 31;
  const int wid  = blockIdx.x * 4 + wave;       // 0..4095
  const int qblk = wid & 127;                   // S/16 query tiles
  const int h    = (wid >> 7) & 15;
  const int b    = wid >> 11;

  const __bf16* qbase  = qws  + (size_t)(b * SEQ + qblk * 16) * DIM + h * HDIM;
  const __bf16* kbase0 = kws  + (size_t)(b * SEQ) * DIM + h * HDIM;
  const __bf16* vbase  = vtws + (size_t)b * DIM * SEQ + (size_t)(h * HDIM) * SEQ;

  const v16bf fa0 = load_frag(qbase, DIM);        // q, K-dims 0..31
  const v16bf fa1 = load_frag(qbase + 32, DIM);   // q, K-dims 32..63

  v8f o[4];
  float mrun[8], lrun[8];
#pragma unroll
  for (int nt = 0; nt < 4; ++nt)
#pragma unroll
    for (int i = 0; i < 8; ++i) o[nt][i] = 0.0f;
#pragma unroll
  for (int v = 0; v < 8; ++v) { mrun[v] = -1e30f; lrun[v] = 0.0f; }

  __bf16* pTile = &ldsP[wave][0];
  const float scale = 0.125f;   // 1/sqrt(64)
  const int NB = SEQ / 32;      // 64 key blocks

  // TDM pipeline prologue: fetch key block 0.
  tdm_load_k_tile(kbase0, &ldsK[wave][0][0]);

  for (int kb = 0; kb < NB; ++kb) {
    // Issue next tile, then wait until the current tile is resident.
    if (kb + 1 < NB) {
      tdm_load_k_tile(kbase0 + (size_t)((kb + 1) * 32) * DIM,
                      &ldsK[wave][(kb + 1) & 1][0]);
      __builtin_amdgcn_s_wait_tensorcnt(1);
    } else {
      __builtin_amdgcn_s_wait_tensorcnt(0);
    }
    const __bf16* kt = &ldsK[wave][kb & 1][0];   // [key][dim], stride 64

    v8f z;
#pragma unroll
    for (int i = 0; i < 8; ++i) z[i] = 0.0f;

    // scores tile 16x32 as two 16x16 C tiles (keys 0-15 / 16-31)
    v16bf fbA0 = load_frag(kt, HDIM);
    v16bf fbA1 = load_frag(kt + 32, HDIM);
    v8f slo = __builtin_amdgcn_wmma_f32_16x16x32_bf16(
        false, fa0, false, fbA0, (short)0, z, false, false);
    slo = __builtin_amdgcn_wmma_f32_16x16x32_bf16(
        false, fa1, false, fbA1, (short)0, slo, false, false);

    v16bf fbB0 = load_frag(kt + 16 * HDIM, HDIM);
    v16bf fbB1 = load_frag(kt + 16 * HDIM + 32, HDIM);
    v8f shi = __builtin_amdgcn_wmma_f32_16x16x32_bf16(
        false, fa0, false, fbB0, (short)0, z, false, false);
    shi = __builtin_amdgcn_wmma_f32_16x16x32_bf16(
        false, fa1, false, fbB1, (short)0, shi, false, false);

    // Online softmax. Row r lives across 16 lanes of one half in VGPR v.
#pragma unroll
    for (int v = 0; v < 8; ++v) {
      float a = slo[v] * scale;
      float c = shi[v] * scale;
      float t = fmaxf(a, c);
#pragma unroll
      for (int msk = 1; msk < 16; msk <<= 1) t = fmaxf(t, __shfl_xor(t, msk, 32));
      const float mnew  = fmaxf(mrun[v], t);
      const float alpha = __expf(mrun[v] - mnew);
      const float pl = __expf(a - mnew);
      const float ph = __expf(c - mnew);
      float rs = pl + ph;
#pragma unroll
      for (int msk = 1; msk < 16; msk <<= 1) rs += __shfl_xor(rs, msk, 32);
      lrun[v] = lrun[v] * alpha + rs;
      mrun[v] = mnew;
#pragma unroll
      for (int nt = 0; nt < 4; ++nt) o[nt][v] *= alpha;
      // C layout -> row-major P tile in LDS (for A-fragment reload)
      const int prow = v + ((lane >> 4) << 3);
      pTile[prow * 32 + (lane & 15)]      = f2bf(pl);
      pTile[prow * 32 + 16 + (lane & 15)] = f2bf(ph);
    }

    // Wave-local LDS write->read ordering (cheaper than a block barrier).
    asm volatile("s_wait_dscnt 0" ::: "memory");

    v16bf fp = load_frag(pTile, 32);                    // P as 16x32 A-frag
    const __bf16* vptr = vbase + kb * 32;
    __builtin_prefetch(vptr + 32, 0, 0);                // next V block
#pragma unroll
    for (int nt = 0; nt < 4; ++nt) {
      v16bf fv = load_frag(vptr + (size_t)(nt * 16) * SEQ, SEQ);
      o[nt] = __builtin_amdgcn_wmma_f32_16x16x32_bf16(
          false, fp, false, fv, (short)0, o[nt], false, false);
    }
  }

  // Normalize and store fp32 output [b][s][h*64+d].
  const int colL    = lane & 15;
  const int rowHalf = (lane >> 4) << 3;
#pragma unroll
  for (int v = 0; v < 8; ++v) {
    const float rl  = 1.0f / lrun[v];
    const int srow  = qblk * 16 + v + rowHalf;
#pragma unroll
    for (int nt = 0; nt < 4; ++nt) {
      out[(size_t)(b * SEQ + srow) * DIM + h * HDIM + nt * 16 + colL] =
          o[nt][v] * rl;
    }
  }
}

extern "C" void kernel_launch(void* const* d_in, const int* in_sizes, int n_in,
                              void* d_out, int out_size, void* d_ws, size_t ws_size,
                              hipStream_t stream) {
  const float* Q  = (const float*)d_in[0];
  const float* K  = (const float*)d_in[1];
  const float* V  = (const float*)d_in[2];
  const float* Wq = (const float*)d_in[3];
  const float* Wk = (const float*)d_in[4];
  const float* Wv = (const float*)d_in[5];
  float* out = (float*)d_out;

  // Workspace: 3 bf16 tensors of [B*S, D] = 3 * 8 MiB = 24 MiB.
  const size_t tEl = (size_t)BATCH * SEQ * DIM;
  __bf16* qws  = (__bf16*)d_ws;
  __bf16* kws  = qws + tEl;
  __bf16* vtws = kws + tEl;

  dim3 gridP(DIM / 64, (BATCH * SEQ) / 128, 3);
  mha_proj_kernel<<<gridP, 256, 0, stream>>>(Q, K, V, Wq, Wk, Wv, qws, kws, vtws);

  const int nWaves = BATCH * NH * (SEQ / 16);   // 4096
  dim3 gridA(nWaves / 4);
  mha_attn_kernel<<<gridA, 128, 0, stream>>>(qws, kws, vtws, out);
}
